// Model_32152125178027
// MI455X (gfx1250) — matrix operand order repaired and checked by reading, not compiled, not run on hardware
//
#include <hip/hip_runtime.h>
#include <hip/hip_bf16.h>

// ---------------------------------------------------------------------------
// CDNA5 (gfx1250) implementation.
//   prep kernels: pre-shuffle weights into WMMA fragment order (f16).
//   conv0..conv3: WMMA f16 GEMM per 64-wide time tile; register-blocked
//                 (1 B-frag reused by 4 m-tile WMMAs).
//   regime scan:  serial in T, wave32 groups-of-8 softmax via shfl_xor.
//   mix GEMM 1/2: stacked-K mixture einsums (K = R*Cin + 8 bias rows).
// ---------------------------------------------------------------------------

typedef __attribute__((ext_vector_type(16))) _Float16 v16h;
typedef __attribute__((ext_vector_type(8)))  float    v8f;

#define B_SZ   32
#define T_SEQ  4096
#define T0_OUT 4110
#define T1_OUT 4108
#define T2_OUT 4104

#define V8F_ZERO {0.f, 0.f, 0.f, 0.f, 0.f, 0.f, 0.f, 0.f}

__device__ __forceinline__ float eluf(float x) {
    return x > 0.f ? x : (__expf(x) - 1.f);
}

// ---------------------------------------------------------------------------
// Fragment-order packers. WMMA 16x16x32 f16 A-fragment packing, contiguous k order as in B reads:
// flat half index = ((mt*KS + ks)*32 + lane)*16 + e, where
//   m  = mt*16 + (lane&15),  half = lane>>4,  vv = e>>1,
//   kb = ks*32 + half*16 + e
// ---------------------------------------------------------------------------
__global__ void prep_conv_frag(const float* __restrict__ w, _Float16* __restrict__ dst,
                               int Cin, int Cout, int MT, int KS) {
    int idx = blockIdx.x * 256 + threadIdx.x;
    int total = MT * KS * 512;
    if (idx >= total) return;
    int e    = idx & 15;
    int lane = (idx >> 4) & 31;
    int tt   = idx >> 9;
    int ks   = tt % KS;
    int mt   = tt / KS;
    int vv   = e >> 1;
    int kb = ks * 32 + ((lane >> 4) << 4) + e;
    int m  = mt * 16 + (lane & 15);
    int k  = (kb >= Cin) ? 1 : 0;       // K ordering: kr = k*Cin + c
    int c  = kb - k * Cin;
    float v = (m < Cout) ? w[((size_t)m * Cin + c) * 2 + k] : 0.f;
    dst[idx] = (_Float16)v;
}

__global__ void prep_mix_frag(const float* __restrict__ pw, const float* __restrict__ pb,
                              _Float16* __restrict__ dst, int Cin, int M, int MT, int KS) {
    int idx = blockIdx.x * 256 + threadIdx.x;
    int total = MT * KS * 512;
    if (idx >= total) return;
    int e    = idx & 15;
    int lane = (idx >> 4) & 31;
    int tt   = idx >> 9;
    int ks   = tt % KS;
    int mt   = tt / KS;
    int vv   = e >> 1;
    int kb = ks * 32 + ((lane >> 4) << 4) + e;
    int m  = mt * 16 + (lane & 15);
    int KW = 8 * Cin;                   // stacked K: kr = r*Cin + c, then 8 bias rows
    float v = 0.f;
    if (kb < KW)            { int r = kb / Cin, c = kb - (kb / Cin) * Cin;
                              v = pw[((size_t)r * M + m) * Cin + c]; }
    else if (kb < KW + 8)   { v = pb[(size_t)(kb - KW) * M + m]; }
    dst[idx] = (_Float16)v;
}

// ---------------------------------------------------------------------------
// Dilated conv (width 2) as WMMA GEMM. B staged once per block in LDS as
// sb[t][kr] (kr = k*CIN + c, padded row stride). Register-blocked: one
// B-frag feeds 4 independent m-tile accumulators per K-step.
// ---------------------------------------------------------------------------
template<int CIN, int DIL, int MT>
__global__ void __launch_bounds__(256, 1)
conv_wmma_kernel(const float* __restrict__ in,
                 const _Float16* __restrict__ wfrag,
                 const float* __restrict__ bias,
                 float* __restrict__ out,
                 int leftPad, int Tsrc, int Tout, int Cout) {
    constexpr int K  = 2 * CIN;
    constexpr int KS = K / 32;
    constexpr int KP = K + 8;           // bank-rotating pad, keeps 16B align
    constexpr int NP = 64 + DIL;
    __shared__ _Float16 sb[64 * KP];

    const int b   = blockIdx.y;
    const int t0  = blockIdx.x * 64;
    const int tid = threadIdx.x;

    // Stage input tile; value feeds tap k=0 at column i and tap k=1 at i-DIL.
    for (int idx = tid; idx < CIN * NP; idx += 256) {
        int c = idx / NP, i = idx - c * NP;
        int ts = t0 + i - leftPad;
        float v = (ts >= 0 && ts < Tsrc) ? in[((size_t)b * CIN + c) * Tsrc + ts] : 0.f;
        _Float16 hv = (_Float16)v;
        if (i < 64) sb[i * KP + c] = hv;
        int t2 = i - DIL;
        if (t2 >= 0) sb[t2 * KP + CIN + c] = hv;
    }
    // Speculative prefetch of the next time tile (global_prefetch_b8).
    __builtin_prefetch(in + ((size_t)b * CIN) * Tsrc + t0 + 64, 0, 1);
    __syncthreads();

    const int wave = tid >> 5, lane = tid & 31;
    const int n = lane & 15, half = lane >> 4;

    if constexpr (MT == 8) {
        const int nt      = wave & 3;
        const int base_mt = (wave >> 2) * 4;          // 0 or 4
        const int ncol    = nt * 16 + n;
        const _Float16* brow  = sb + ncol * KP + half * 16;
        const _Float16* abase = wfrag + (size_t)base_mt * KS * 512 + lane * 16;
        v8f acc[4] = {V8F_ZERO, V8F_ZERO, V8F_ZERO, V8F_ZERO};
        #pragma unroll 2
        for (int ks = 0; ks < KS; ++ks) {
            v16h bf = *(const v16h*)(brow + ks * 32);
            const _Float16* ap = abase + ks * 512;
            #pragma unroll
            for (int q = 0; q < 4; ++q) {
                v16h a = *(const v16h*)(ap + q * (KS * 512));
                acc[q] = __builtin_amdgcn_wmma_f32_16x16x32_f16(
                    false, a, false, bf, (short)0, acc[q], false, false);
            }
        }
        const int t = t0 + nt * 16 + n;
        if (t < Tout) {
            #pragma unroll
            for (int q = 0; q < 4; ++q) {
                #pragma unroll
                for (int vv = 0; vv < 8; ++vv) {
                    int o = (base_mt + q) * 16 + vv + half * 8;   // Cout==128 here
                    out[((size_t)b * 128 + o) * Tout + t] = eluf(acc[q][vv] + bias[o]);
                }
            }
        }
    } else {   // MT == 1 (conv3: Cout = 9, waves 0..3 each take one n-tile)
        if (wave < 4) {
            const int nt   = wave;
            const int ncol = nt * 16 + n;
            const _Float16* brow  = sb + ncol * KP + half * 16;
            const _Float16* abase = wfrag + lane * 16;
            v8f a0 = V8F_ZERO, a1 = V8F_ZERO;
            #pragma unroll
            for (int ks = 0; ks < KS; ks += 2) {
                v16h bf0 = *(const v16h*)(brow + ks * 32);
                v16h aa0 = *(const v16h*)(abase + ks * 512);
                a0 = __builtin_amdgcn_wmma_f32_16x16x32_f16(
                    false, aa0, false, bf0, (short)0, a0, false, false);
                v16h bf1 = *(const v16h*)(brow + (ks + 1) * 32);
                v16h aa1 = *(const v16h*)(abase + (ks + 1) * 512);
                a1 = __builtin_amdgcn_wmma_f32_16x16x32_f16(
                    false, aa1, false, bf1, (short)0, a1, false, false);
            }
            v8f acc = a0 + a1;
            const int t = t0 + nt * 16 + n;
            #pragma unroll
            for (int vv = 0; vv < 8; ++vv) {
                int o = vv + half * 8;
                if (o < Cout && t < Tout)
                    out[((size_t)b * Cout + o) * Tout + t] = eluf(acc[vv] + bias[o]);
            }
        }
    }
}

// ---------------------------------------------------------------------------
// Regime dynamics scan: grid=8 x 32 threads; lane = (batchGroup<<3)|regime.
// ---------------------------------------------------------------------------
__device__ __forceinline__ float softmax8(float logit) {
    float mx = logit;
    mx = fmaxf(mx, __shfl_xor(mx, 1, 8));
    mx = fmaxf(mx, __shfl_xor(mx, 2, 8));
    mx = fmaxf(mx, __shfl_xor(mx, 4, 8));
    float e = __expf(logit - mx);
    float s = e;
    s += __shfl_xor(s, 1, 8);
    s += __shfl_xor(s, 2, 8);
    s += __shfl_xor(s, 4, 8);
    return e / s;
}

__global__ void regime_scan_kernel(const float* __restrict__ h,
                                   const float* __restrict__ gu,
                                   float* __restrict__ zbuf,
                                   float* __restrict__ zout) {
    int lane = threadIdx.x;
    int grp = lane >> 3, r = lane & 7;
    int b = blockIdx.x * 4 + grp;
    const float* hr  = h  + ((size_t)b * 9 + r) * T_SEQ;
    const float* h8  = h  + ((size_t)b * 9 + 8) * T_SEQ;
    const float* gur = gu + ((size_t)b * 8 + r) * T_SEQ;
    float* zb = zbuf + ((size_t)b * 8 + r) * T_SEQ;
    float* zo = zout + ((size_t)b * 8 + r) * T_SEQ;

    float zprev = 0.f;
    for (int t = 0; t < T_SEQ; ++t) {
        float p = softmax8(hr[t]);
        float g = -__logf(-__logf(gur[t]));
        float np;
        if (t == 0) {
            np = p;
        } else {
            float it = 0.5f + 0.5f * tanhf(h8[t]);
            it = fminf(fmaxf(it, 0.01f), 0.99f);
            np = (1.f - it) * p + it * zprev;
        }
        float z = softmax8((__logf(np + 1e-6f) + g) * 5.0f);  // 1/GEN_TEMP
        zb[t] = z;
        zo[t] = z;
        zprev = z;
    }
}

// ---------------------------------------------------------------------------
// Mix GEMM 1: params1[b,e,t] = sum_r z*(pw0[r]@pre_h + pb0[r]).
// K = 8*128 + 8 bias rows -> 33 k-steps. Each wave: 4 m-tiles x 1 n-tile,
// one B-frag reused per K-step across 4 accumulators.
// ---------------------------------------------------------------------------
__global__ void __launch_bounds__(256, 1)
mix_gemm1_kernel(const float* __restrict__ x3,
                 const float* __restrict__ zbuf,
                 const _Float16* __restrict__ wfrag,
                 float* __restrict__ out) {
    constexpr int KF = 1056, KP = KF + 8, KS = 33;
    extern __shared__ char smem[];
    _Float16* sb   = (_Float16*)smem;            // [32][KP]
    _Float16* sx16 = sb + 32 * KP;               // [128][32]
    float*    sz   = (float*)(sx16 + 128 * 32);  // [8][32]
    const int b = blockIdx.y, t0 = blockIdx.x * 32, tid = threadIdx.x;

    for (int idx = tid; idx < 128 * 32; idx += 256) {
        int j = idx >> 5, nn = idx & 31;
        sx16[idx] = (_Float16)x3[((size_t)b * 128 + j) * T2_OUT + (t0 + nn) + 8]; // pre_h
    }
    for (int idx = tid; idx < 8 * 32; idx += 256) {
        int r = idx >> 5, nn = idx & 31;
        sz[idx] = zbuf[((size_t)b * 8 + r) * T_SEQ + t0 + nn];
    }
    __syncthreads();
    for (int idx = tid; idx < 32 * KF; idx += 256) {
        int kr = idx >> 5, nn = idx & 31;
        float v;
        if (kr < 1024)      { int r = kr >> 7, c = kr & 127;
                              v = sz[r * 32 + nn] * (float)sx16[c * 32 + nn]; }
        else if (kr < 1032) { v = sz[(kr - 1024) * 32 + nn]; }
        else                { v = 0.f; }
        sb[nn * KP + kr] = (_Float16)v;
    }
    __syncthreads();

    const int wave = tid >> 5, lane = tid & 31;
    const int n = lane & 15, half = lane >> 4;
    const int nt      = wave & 1;
    const int base_mt = (wave >> 1) * 4;            // 0,4,8,12
    const _Float16* brow  = sb + (nt * 16 + n) * KP + half * 16;
    const _Float16* abase = wfrag + (size_t)base_mt * KS * 512 + lane * 16;

    v8f acc[4] = {V8F_ZERO, V8F_ZERO, V8F_ZERO, V8F_ZERO};
    #pragma unroll 1
    for (int ks = 0; ks < KS; ++ks) {
        v16h bf = *(const v16h*)(brow + ks * 32);
        const _Float16* ap = abase + ks * 512;
        #pragma unroll
        for (int q = 0; q < 4; ++q) {
            v16h a = *(const v16h*)(ap + q * (KS * 512));
            acc[q] = __builtin_amdgcn_wmma_f32_16x16x32_f16(
                false, a, false, bf, (short)0, acc[q], false, false);
        }
    }
    const int t = t0 + nt * 16 + n;
    #pragma unroll
    for (int q = 0; q < 4; ++q) {
        #pragma unroll
        for (int vv = 0; vv < 8; ++vv) {
            int e = (base_mt + q) * 16 + vv + half * 8;
            out[((size_t)b * 256 + e) * T_SEQ + t] = acc[q][vv];
        }
    }
}

// ---------------------------------------------------------------------------
// Mix GEMM 2: K = 8*256 + 8 -> 65 k-steps, M=64 (4 m-tiles x 2 n-tiles ->
// one tile per wave, dual accumulators). Splits into mus / clipped logvars.
// ---------------------------------------------------------------------------
__global__ void __launch_bounds__(256, 1)
mix_gemm2_kernel(const float* __restrict__ p1,
                 const float* __restrict__ zbuf,
                 const _Float16* __restrict__ wfrag,
                 float* __restrict__ mus,
                 float* __restrict__ logv) {
    constexpr int KF = 2080, KP = KF + 8, KS = 65;
    extern __shared__ char smem[];
    _Float16* sb   = (_Float16*)smem;            // [32][KP]
    _Float16* sp16 = sb + 32 * KP;               // [256][32]
    float*    sz   = (float*)(sp16 + 256 * 32);  // [8][32]
    const int b = blockIdx.y, t0 = blockIdx.x * 32, tid = threadIdx.x;

    for (int idx = tid; idx < 256 * 32; idx += 256) {
        int j = idx >> 5, nn = idx & 31;
        sp16[idx] = (_Float16)p1[((size_t)b * 256 + j) * T_SEQ + t0 + nn];
    }
    for (int idx = tid; idx < 8 * 32; idx += 256) {
        int r = idx >> 5, nn = idx & 31;
        sz[idx] = zbuf[((size_t)b * 8 + r) * T_SEQ + t0 + nn];
    }
    __syncthreads();
    for (int idx = tid; idx < 32 * KF; idx += 256) {
        int kr = idx >> 5, nn = idx & 31;
        float v;
        if (kr < 2048)      { int r = kr >> 8, c = kr & 255;
                              v = sz[r * 32 + nn] * (float)sp16[c * 32 + nn]; }
        else if (kr < 2056) { v = sz[(kr - 2048) * 32 + nn]; }
        else                { v = 0.f; }
        sb[nn * KP + kr] = (_Float16)v;
    }
    __syncthreads();

    const int wave = tid >> 5, lane = tid & 31;
    const int n = lane & 15, half = lane >> 4;
    const int mt = wave >> 1, nt = wave & 1;        // 8 tiles, one per wave
    const _Float16* brow  = sb + (nt * 16 + n) * KP + half * 16;
    const _Float16* abase = wfrag + (size_t)mt * KS * 512 + lane * 16;

    v8f a0 = V8F_ZERO, a1 = V8F_ZERO;
    #pragma unroll 1
    for (int ks = 0; ks < KS - 1; ks += 2) {
        v16h bf0 = *(const v16h*)(brow + ks * 32);
        v16h aa0 = *(const v16h*)(abase + ks * 512);
        a0 = __builtin_amdgcn_wmma_f32_16x16x32_f16(
            false, aa0, false, bf0, (short)0, a0, false, false);
        v16h bf1 = *(const v16h*)(brow + (ks + 1) * 32);
        v16h aa1 = *(const v16h*)(abase + (ks + 1) * 512);
        a1 = __builtin_amdgcn_wmma_f32_16x16x32_f16(
            false, aa1, false, bf1, (short)0, a1, false, false);
    }
    {   // tail k-step (ks = 64: bias rows)
        v16h bf = *(const v16h*)(brow + (KS - 1) * 32);
        v16h aa = *(const v16h*)(abase + (KS - 1) * 512);
        a0 = __builtin_amdgcn_wmma_f32_16x16x32_f16(
            false, aa, false, bf, (short)0, a0, false, false);
    }
    v8f acc = a0 + a1;
    const int t = t0 + nt * 16 + n;
    #pragma unroll
    for (int vv = 0; vv < 8; ++vv) {
        int e = mt * 16 + vv + half * 8;
        float y = acc[vv];
        if (e < 32) {
            mus[((size_t)b * 32 + e) * T_SEQ + t] = y;
        } else {
            y = fminf(fmaxf(y, -3.f), 1.f);
            logv[((size_t)b * 32 + (e - 32)) * T_SEQ + t] = y;
        }
    }
}

// ---------------------------------------------------------------------------
// Scratch layout (bytes, all 256-aligned). params1 aliases dead x1+x2 region.
// ---------------------------------------------------------------------------
static const size_t SZ_X1  = (size_t)B_SZ * 128 * T0_OUT * 4;   // 67,338,240
static const size_t SZ_X2  = (size_t)B_SZ * 128 * T1_OUT * 4;   // 67,305,472
static const size_t SZ_X3  = (size_t)B_SZ * 128 * T2_OUT * 4;   // 67,239,936
static const size_t SZ_H   = (size_t)B_SZ * 9 * T_SEQ * 4;
static const size_t SZ_Z   = (size_t)B_SZ * 8 * T_SEQ * 4;
static const size_t SZ_CA0 = (size_t)8 * 2 * 512 * 2;           // conv0 frags
static const size_t SZ_CA1 = (size_t)8 * 8 * 512 * 2;           // conv1 frags
static const size_t SZ_CA2 = (size_t)8 * 8 * 512 * 2;           // conv2 frags
static const size_t SZ_CA3 = (size_t)1 * 8 * 512 * 2;           // conv3 frags
static const size_t SZ_W1  = (size_t)16 * 33 * 512 * 2;         // mix1 frags
static const size_t OFF_X1  = 0;
static const size_t OFF_X2  = OFF_X1 + SZ_X1;
static const size_t OFF_X3  = OFF_X2 + SZ_X2;
static const size_t OFF_H   = OFF_X3 + SZ_X3;
static const size_t OFF_Z   = OFF_H + SZ_H;
static const size_t OFF_CA0 = OFF_Z + SZ_Z;
static const size_t OFF_CA1 = OFF_CA0 + SZ_CA0;
static const size_t OFF_CA2 = OFF_CA1 + SZ_CA1;
static const size_t OFF_CA3 = OFF_CA2 + SZ_CA2;
static const size_t OFF_W1  = OFF_CA3 + SZ_CA3;
static const size_t OFF_W2  = OFF_W1 + SZ_W1;
static const size_t OFF_P1  = 0;   // alias over x1+x2 (134.6 MB >= 134.2 MB)

extern "C" void kernel_launch(void* const* d_in, const int* in_sizes, int n_in,
                              void* d_out, int out_size, void* d_ws, size_t ws_size,
                              hipStream_t stream) {
    const float* x   = (const float*)d_in[0];
    const float* w0  = (const float*)d_in[1];
    const float* b0  = (const float*)d_in[2];
    const float* w1  = (const float*)d_in[3];
    const float* b1  = (const float*)d_in[4];
    const float* w2  = (const float*)d_in[5];
    const float* b2  = (const float*)d_in[6];
    const float* w3  = (const float*)d_in[7];
    const float* b3  = (const float*)d_in[8];
    const float* pw0 = (const float*)d_in[9];
    const float* pb0 = (const float*)d_in[10];
    const float* pw1 = (const float*)d_in[11];
    const float* pb1 = (const float*)d_in[12];
    const float* gu  = (const float*)d_in[13];

    char* ws = (char*)d_ws;
    float*    x1  = (float*)(ws + OFF_X1);
    float*    x2  = (float*)(ws + OFF_X2);
    float*    x3  = (float*)(ws + OFF_X3);
    float*    hb  = (float*)(ws + OFF_H);
    float*    zb  = (float*)(ws + OFF_Z);
    _Float16* cA0 = (_Float16*)(ws + OFF_CA0);
    _Float16* cA1 = (_Float16*)(ws + OFF_CA1);
    _Float16* cA2 = (_Float16*)(ws + OFF_CA2);
    _Float16* cA3 = (_Float16*)(ws + OFF_CA3);
    _Float16* wA1 = (_Float16*)(ws + OFF_W1);
    _Float16* wA2 = (_Float16*)(ws + OFF_W2);
    float*    p1  = (float*)(ws + OFF_P1);

    float* out   = (float*)d_out;
    float* musO  = out;
    float* logvO = out + (size_t)B_SZ * 32 * T_SEQ;
    float* zO    = out + (size_t)2 * B_SZ * 32 * T_SEQ;

    dim3 blk(256);

    // ---- weight fragment packers (one-time, tiny) ----
    prep_conv_frag<<<(8 * 2 * 512 + 255) / 256, blk, 0, stream>>>(w0, cA0, 32, 128, 8, 2);
    prep_conv_frag<<<(8 * 8 * 512 + 255) / 256, blk, 0, stream>>>(w1, cA1, 128, 128, 8, 8);
    prep_conv_frag<<<(8 * 8 * 512 + 255) / 256, blk, 0, stream>>>(w2, cA2, 128, 128, 8, 8);
    prep_conv_frag<<<(1 * 8 * 512 + 255) / 256, blk, 0, stream>>>(w3, cA3, 128, 9, 1, 8);
    prep_mix_frag<<<(16 * 33 * 512 + 255) / 256, blk, 0, stream>>>(pw0, pb0, wA1, 128, 256, 16, 33);
    prep_mix_frag<<<(4 * 65 * 512 + 255) / 256, blk, 0, stream>>>(pw1, pb1, wA2, 256, 64, 4, 65);

    // ---- conv stack (left-pad 15 folded into conv0) ----
    conv_wmma_kernel<32, 1, 8><<<dim3(65, B_SZ), blk, 0, stream>>>(
        x, cA0, b0, x1, 15, T_SEQ, T0_OUT, 128);
    conv_wmma_kernel<128, 2, 8><<<dim3(65, B_SZ), blk, 0, stream>>>(
        x1, cA1, b1, x2, 0, T0_OUT, T1_OUT, 128);
    conv_wmma_kernel<128, 4, 8><<<dim3(65, B_SZ), blk, 0, stream>>>(
        x2, cA2, b2, x3, 0, T1_OUT, T2_OUT, 128);
    conv_wmma_kernel<128, 8, 1><<<dim3(64, B_SZ), blk, 0, stream>>>(
        x3, cA3, b3, hb, 0, T2_OUT, T_SEQ, 9);

    // ---- sequential regime scan (also writes z into d_out) ----
    regime_scan_kernel<<<8, 32, 0, stream>>>(hb, gu, zb, zO);

    // ---- mixture GEMMs ----
    {
        size_t sh = (size_t)(32 * (1056 + 8) + 128 * 32) * 2 + 8 * 32 * 4;
        mix_gemm1_kernel<<<dim3(T_SEQ / 32, B_SZ), blk, sh, stream>>>(x3, zb, wA1, p1);
    }
    {
        size_t sh = (size_t)(32 * (2080 + 8) + 256 * 32) * 2 + 8 * 32 * 4;
        mix_gemm2_kernel<<<dim3(T_SEQ / 32, B_SZ), blk, sh, stream>>>(p1, zb, wA2, musO, logvO);
    }
}
